// MoEModule_1443109011473
// MI455X (gfx1250) — matrix operand order repaired and checked by reading, not compiled, not run on hardware
//
#include <hip/hip_runtime.h>
#include <hip/hip_bf16.h>
#include <math.h>

#define B_ 8
#define S_ 4096
#define T_ (B_*S_)        // 32768 tokens
#define D_ 512
#define F_ 1024
#define E_ 8
#define K_ 2

typedef __attribute__((ext_vector_type(16))) __bf16 v16bf;
typedef __attribute__((ext_vector_type(8)))  float  v8f;

union BF16x16 { unsigned short u[16]; v16bf v; };

__device__ __forceinline__ unsigned short f2bf(float f) {
  unsigned int u = __float_as_uint(f);
  u += 0x7FFFu + ((u >> 16) & 1u);          // round-to-nearest-even
  return (unsigned short)(u >> 16);
}

__device__ __forceinline__ void gatomic_add(float* p, float v) {
  // relaxed, device-scope fp32 add -> native global_atomic_add_f32
  __hip_atomic_fetch_add(p, v, __ATOMIC_RELAXED, __HIP_MEMORY_SCOPE_AGENT);
}

// ---------------------------------------------------------------- init
__global__ void k_init(float* out, long long nout, int* assign, int nassign,
                       int* counts, int* fills, float* ssum) {
  long long i = (long long)blockIdx.x * blockDim.x + threadIdx.x;
  long long stride = (long long)gridDim.x * blockDim.x;
  for (long long j = i; j < nout; j += stride) out[j] = 0.f;
  for (long long j = i; j < nassign; j += stride) assign[j] = -1;
  if (i < E_) { counts[i] = 0; fills[i] = 0; ssum[i] = 0.f; }
}

// ---------------------------------------------------------- f32 -> bf16
__global__ void k_cvt_bf16(const float* __restrict__ src,
                           unsigned short* __restrict__ dst, int n) {
  int i = blockIdx.x * blockDim.x + threadIdx.x;
  int stride = gridDim.x * blockDim.x;
  for (int j = i; j < n; j += stride) dst[j] = f2bf(src[j]);
}

// ---------------------------------------------------------------- gate
__global__ void k_gate(const float* __restrict__ x, const float* __restrict__ gate_w,
                       int* __restrict__ sel, int* __restrict__ counts,
                       float* __restrict__ ssum) {
  __shared__ float gw[E_ * D_];   // 16 KB
  __shared__ float ls[E_];
  __shared__ int   lc[E_];
  int tid = threadIdx.x;
  for (int i = tid; i < E_ * D_; i += blockDim.x) gw[i] = gate_w[i];
  if (tid < E_) { ls[tid] = 0.f; lc[tid] = 0; }
  __syncthreads();

  int t = blockIdx.x * blockDim.x + tid;
  float acc[E_];
  #pragma unroll
  for (int e = 0; e < E_; e++) acc[e] = 0.f;
  const float4* xr = (const float4*)(x + (size_t)t * D_);
  for (int i = 0; i < D_ / 4; i++) {
    float4 v = xr[i];
    #pragma unroll
    for (int e = 0; e < E_; e++) {
      const float* g = &gw[e * D_ + 4 * i];
      acc[e] += v.x * g[0] + v.y * g[1] + v.z * g[2] + v.w * g[3];
    }
  }
  // softmax over 8
  float m = acc[0];
  #pragma unroll
  for (int e = 1; e < E_; e++) m = fmaxf(m, acc[e]);
  float s = 0.f, p[E_];
  #pragma unroll
  for (int e = 0; e < E_; e++) { p[e] = __expf(acc[e] - m); s += p[e]; }
  float inv = __builtin_amdgcn_rcpf(s);
  int e0 = 0; float b0 = -1.f;
  #pragma unroll
  for (int e = 0; e < E_; e++) { p[e] *= inv; if (p[e] > b0) { b0 = p[e]; e0 = e; } }
  int e1 = 0; float b1 = -1.f;
  #pragma unroll
  for (int e = 0; e < E_; e++) { if (e != e0 && p[e] > b1) { b1 = p[e]; e1 = e; } }
  sel[t] = e0 | (e1 << 8);

  #pragma unroll
  for (int e = 0; e < E_; e++) atomicAdd(&ls[e], p[e]);
  atomicAdd(&lc[e0], 1); atomicAdd(&lc[e1], 1);
  __syncthreads();
  if (tid < E_) { atomicAdd(&ssum[tid], ls[tid]); atomicAdd(&counts[tid], lc[tid]); }
}

// ------------------------------------------------- offsets + lb_loss
__global__ void k_offsets(const int* __restrict__ counts, const float* __restrict__ ssum,
                          int* __restrict__ offs, float* __restrict__ loss_out) {
  if (threadIdx.x == 0 && blockIdx.x == 0) {
    int off = 0;
    for (int e = 0; e < E_; e++) { offs[e] = off; off += (counts[e] + 15) & ~15; }
    offs[E_] = off;
    float loss = 0.f;
    for (int e = 0; e < E_; e++) {
      float pr = ssum[e] / (float)T_;
      loss += pr * logf(pr + 1e-8f);
    }
    *loss_out = loss;
  }
}

// -------------------------------------------------------------- scatter
__global__ void k_scatter(const int* __restrict__ sel, const int* __restrict__ offs,
                          int* __restrict__ fills, int* __restrict__ assign) {
  int t = blockIdx.x * blockDim.x + threadIdx.x;
  int sv = sel[t];
  int e0 = sv & 255, e1 = (sv >> 8) & 255;
  int p0 = offs[e0] + atomicAdd(&fills[e0], 1);
  assign[p0] = t;
  int p1 = offs[e1] + atomicAdd(&fills[e1], 1);
  assign[p1] = t;
}

// ----------------------------------------------------- fused expert FFN
#define XSTR (D_ + 8)    // ushort stride, 16B-aligned rows
#define HSTR (F_ + 8)

__global__ __launch_bounds__(256) void k_moe_gemm(
    const float* __restrict__ x,
    const unsigned short* __restrict__ w1b,   // [E][F][D] bf16
    const unsigned short* __restrict__ w3b,   // [E][F][D] bf16
    const unsigned short* __restrict__ w2b,   // [E][D][F] bf16
    const int* __restrict__ assign,
    const int* __restrict__ offs,
    float* __restrict__ out)
{
  __shared__ unsigned short Xs[16 * XSTR];    // 16.6 KB bf16 X tile
  __shared__ unsigned short Hs[16 * HSTR];    // 33.0 KB bf16 h tile
  __shared__ int tok[16];

  int base = blockIdx.x * 16;
  int total = offs[E_];
  if (base >= total) return;

  int eid = 0;
  #pragma unroll
  for (int e = 1; e < E_; e++) if (base >= offs[e]) eid = e;

  int tid = threadIdx.x;
  if (tid < 16) tok[tid] = assign[base + tid];
  __syncthreads();

  // ---- stage gathered X rows into LDS as bf16 ----
  {
    int row = tid >> 4;           // 0..15
    int col = (tid & 15) * 32;    // 32 elems per thread
    int t = tok[row];
    if (t >= 0) {
      const float4* xr = (const float4*)(x + (size_t)t * D_ + col);
      #pragma unroll
      for (int j = 0; j < 8; j++) {
        float4 v = xr[j];
        Xs[row * XSTR + col + 4 * j + 0] = f2bf(v.x);
        Xs[row * XSTR + col + 4 * j + 1] = f2bf(v.y);
        Xs[row * XSTR + col + 4 * j + 2] = f2bf(v.z);
        Xs[row * XSTR + col + 4 * j + 3] = f2bf(v.w);
      }
    } else {
      #pragma unroll
      for (int j = 0; j < 32; j++) Xs[row * XSTR + col + j] = 0;
    }
  }
  __syncthreads();

  int wave = tid >> 5;
  int lane = tid & 31;
  int half = lane >> 4;     // A/B K-half select (this code's contiguous order: k = 16*half + s)
  int n16  = lane & 15;     // A row (M) / B,C,D column (N)

  const int    f0w = wave * 128;               // this wave's F slice
  const int    d0  = wave * 64;                // this wave's D slice (phase 2)
  const size_t eW  = (size_t)eid * F_ * D_;
  const size_t eW2 = (size_t)eid * D_ * F_;
  const unsigned short* w2pf = w2b + eW2 + (size_t)d0 * F_;  // prefetch base

  // ---- phase 1: h = silu(X w1^T) * (X w3^T), per-wave 16x128 slice ----
  #pragma unroll
  for (int g = 0; g < 2; g++) {                // two groups of 4 N-tiles
    v8f acc1[4], acc3[4];
    #pragma unroll
    for (int i = 0; i < 4; i++) {
      acc1[i] = (v8f){0.f,0.f,0.f,0.f,0.f,0.f,0.f,0.f};
      acc3[i] = (v8f){0.f,0.f,0.f,0.f,0.f,0.f,0.f,0.f};
    }
    for (int kc = 0; kc < D_; kc += 32) {
      BF16x16 a;
      const unsigned short* ap = &Xs[n16 * XSTR + kc + half * 16];
      #pragma unroll
      for (int i = 0; i < 8; i++) a.u[i]     = ap[i];       // K lo group
      #pragma unroll
      for (int i = 0; i < 8; i++) a.u[8 + i] = ap[8 + i];  // K hi group
      #pragma unroll
      for (int nt = 0; nt < 4; nt++) {
        int f = f0w + g * 64 + nt * 16 + n16;
        const unsigned short* bp1 = w1b + eW + (size_t)f * D_ + kc + half * 16;
        const unsigned short* bp3 = w3b + eW + (size_t)f * D_ + kc + half * 16;
        BF16x16 b1, b3;
        #pragma unroll
        for (int i = 0; i < 16; i++) b1.u[i] = bp1[i];
        #pragma unroll
        for (int i = 0; i < 16; i++) b3.u[i] = bp3[i];
        acc1[nt] = __builtin_amdgcn_wmma_f32_16x16x32_bf16(
            false, a.v, false, b1.v, (short)0, acc1[nt], false, false);
        acc3[nt] = __builtin_amdgcn_wmma_f32_16x16x32_bf16(
            false, a.v, false, b3.v, (short)0, acc3[nt], false, false);
      }
      // spread prefetch of this wave's 128KB w2 slice (phase-2 B operand)
      // across phase 1: 32 slots x 32 lanes x 128B = 128KB
      {
        int slot = g * 16 + (kc >> 5);
        __builtin_prefetch(w2pf + ((size_t)(slot * 32 + lane)) * 64, 0, 3);
      }
    }
    // SwiGLU + store h slice to LDS (C/D layout: row = half*8+r, col = n16)
    #pragma unroll
    for (int nt = 0; nt < 4; nt++) {
      int fcol = f0w + g * 64 + nt * 16 + n16;
      #pragma unroll
      for (int r = 0; r < 8; r++) {
        float a1 = acc1[nt][r];
        float sg = a1 * __builtin_amdgcn_rcpf(1.f + __expf(-a1));  // silu via v_rcp
        Hs[(half * 8 + r) * HSTR + fcol] = f2bf(sg * acc3[nt][r]);
      }
    }
  }
  __syncthreads();

  // ---- phase 2: out_tile = h @ w2^T, per-wave 16x64 slice of D ----
  v8f acc[4];
  #pragma unroll
  for (int i = 0; i < 4; i++) acc[i] = (v8f){0.f,0.f,0.f,0.f,0.f,0.f,0.f,0.f};
  for (int kc = 0; kc < F_; kc += 32) {
    BF16x16 a;
    const unsigned short* ap = &Hs[n16 * HSTR + kc + half * 16];
    #pragma unroll
    for (int i = 0; i < 8; i++) a.u[i]     = ap[i];
    #pragma unroll
    for (int i = 0; i < 8; i++) a.u[8 + i] = ap[8 + i];
    #pragma unroll
    for (int nt = 0; nt < 4; nt++) {
      int d = d0 + nt * 16 + n16;
      const unsigned short* bp = w2b + eW2 + (size_t)d * F_ + kc + half * 16;
      BF16x16 b;
      #pragma unroll
      for (int i = 0; i < 16; i++) b.u[i] = bp[i];
      acc[nt] = __builtin_amdgcn_wmma_f32_16x16x32_bf16(
          false, a.v, false, b.v, (short)0, acc[nt], false, false);
    }
  }
  // scatter-add: each token appears in exactly 2 expert tiles -> atomics
  #pragma unroll
  for (int nt = 0; nt < 4; nt++) {
    int dcol = d0 + nt * 16 + n16;
    #pragma unroll
    for (int r = 0; r < 8; r++) {
      int mrow = half * 8 + r;
      int t = tok[mrow];
      if (t >= 0) gatomic_add(&out[(size_t)t * D_ + dcol], acc[nt][r]);
    }
  }
}

// ---------------------------------------------------------------- launch
extern "C" void kernel_launch(void* const* d_in, const int* in_sizes, int n_in,
                              void* d_out, int out_size, void* d_ws, size_t ws_size,
                              hipStream_t stream) {
  const float* x      = (const float*)d_in[0];
  const float* gate_w = (const float*)d_in[1];
  const float* w1     = (const float*)d_in[2];
  const float* w2     = (const float*)d_in[3];   // note: w2 before w3 in dict order
  const float* w3     = (const float*)d_in[4];
  float* out = (float*)d_out;

  char* ws = (char*)d_ws;
  unsigned short* w1b = (unsigned short*)(ws + 0);
  unsigned short* w3b = (unsigned short*)(ws + 8388608);
  unsigned short* w2b = (unsigned short*)(ws + 16777216);
  char* ctrl = ws + 25165824;
  int*   counts = (int*)(ctrl + 0);
  int*   fills  = (int*)(ctrl + 32);
  int*   offs   = (int*)(ctrl + 64);     // 9 ints
  float* ssum   = (float*)(ctrl + 128);
  int*   sel    = (int*)(ctrl + 256);               // T ints
  int*   assign = (int*)(ctrl + 256 + T_ * 4);      // T*K + E*16 ints

  const int NASSIGN = T_ * K_ + E_ * 16;            // 65664 (pad per expert)
  const long long NOUT = (long long)T_ * D_ + 1;    // output + lb_loss

  k_init   <<<2048, 256, 0, stream>>>(out, NOUT, assign, NASSIGN, counts, fills, ssum);
  k_cvt_bf16<<<2048, 256, 0, stream>>>(w1, w1b, E_ * F_ * D_);
  k_cvt_bf16<<<2048, 256, 0, stream>>>(w3, w3b, E_ * F_ * D_);
  k_cvt_bf16<<<2048, 256, 0, stream>>>(w2, w2b, E_ * D_ * F_);
  k_gate   <<<T_ / 256, 256, 0, stream>>>(x, gate_w, sel, counts, ssum);
  k_offsets<<<1, 32, 0, stream>>>(counts, ssum, offs, out + (size_t)T_ * D_);
  k_scatter<<<T_ / 256, 256, 0, stream>>>(sel, offs, fills, assign);
  k_moe_gemm<<<NASSIGN / 16, 256, 0, stream>>>(x, w1b, w3b, w2b, assign, offs, out);
}
